// NeuralODE_80453327389239
// MI455X (gfx1250) — matrix operand order repaired and checked by reading, not compiled, not run on hardware
//
#include <hip/hip_runtime.h>
#include <math.h>

typedef __attribute__((ext_vector_type(16))) __bf16 v16bf;
typedef __attribute__((ext_vector_type(8)))  __bf16 v8bf;
typedef __attribute__((ext_vector_type(8)))  float  v8f;

#define ASTRIDE 136      // ushorts per staging row: 128 + 8 pad (spreads LDS banks)
#define MDIM    128
#define INDIM   64
#define BATCH   8192
#define TSTEPS  48
#define HSTEP   0.05f
#define NWAVES  4        // waves per block, feature dim split 4 ways
#define TPW     2        // 16-col N-tiles owned per wave (8 tiles total)

__device__ __forceinline__ unsigned short f32_bf16(float f) {
  unsigned int u = __float_as_uint(f);
  u += 0x7FFFu + ((u >> 16) & 1u);           // round-to-nearest-even
  return (unsigned short)(u >> 16);
}

__device__ __forceinline__ float fast_tanh(float x) {
#if __has_builtin(__builtin_amdgcn_tanhf)
  return __builtin_amdgcn_tanhf(x);          // gfx1250 v_tanh_f32
#else
  return tanhf(x);
#endif
}

// Workgroup barrier with LDS-only (address-space-qualified) fences: emits just
// the s_wait_dscnt needed for LDS visibility + s_barrier_signal/wait, does NOT
// drain outstanding global stores (x_transformed writes stay in the background)
// and does NOT clobber global memory, so loop-invariant weight fragments remain
// hoisted in registers across the whole time loop.
__device__ __forceinline__ void lds_barrier() {
  __builtin_amdgcn_fence(__ATOMIC_RELEASE, "workgroup", "local");
  __builtin_amdgcn_s_barrier();
  __builtin_amdgcn_fence(__ATOMIC_ACQUIRE, "workgroup", "local");
}

// Load one 16x32 bf16 A-fragment for this lane from the LDS staging buffer.
// A layout (matches B): lane (m = lane%16, g = lane/16); elements 0..15 hold
// K = 16g..16g+15 (within the K-block).
__device__ __forceinline__ v16bf load_afrag(const unsigned short* arow, int kb, int g) {
  v8bf lo = *(const v8bf*)(arow + kb * 32 + g * 16);
  v8bf hi = *(const v8bf*)(arow + kb * 32 + g * 16 + 8);
  v16bf r;
#pragma unroll
  for (int i = 0; i < 8; ++i) { r[i] = lo[i]; r[8 + i] = hi[i]; }
  return r;
}

// acc[j] (16x16 D-layout, j=0..TPW-1, N-tiles tb..tb+TPW-1) =
//   A(16xK, LDS staging) x B(K x 16*TPW slice of weights).
// Bmat rows are K-contiguous (W row-major == W^T as B; U^T row-major == U as B).
template <int NKB, int BROW>
__device__ __forceinline__ void gemm_tiles(v8f acc[TPW], const unsigned short* Abuf,
                                           const unsigned short* __restrict__ Bmat,
                                           int tb, int n, int g) {
  v16bf af[NKB];
  const unsigned short* arow = Abuf + n * ASTRIDE;
#pragma unroll
  for (int kb = 0; kb < NKB; ++kb) af[kb] = load_afrag(arow, kb, g);
#pragma unroll
  for (int j = 0; j < TPW; ++j) {
    v8f c = {0.f, 0.f, 0.f, 0.f, 0.f, 0.f, 0.f, 0.f};
    const unsigned short* brow = Bmat + (size_t)((tb + j) * 16 + n) * BROW + g * 16;
#pragma unroll
    for (int kb = 0; kb < NKB; ++kb) {
      v16bf bf = *(const v16bf*)(brow + kb * 32);
      c = __builtin_amdgcn_wmma_f32_16x16x32_bf16(false, af[kb], false, bf,
                                                  (short)0, c, false, false);
    }
    acc[j] = c;
  }
}

// One-shot weight conversion: W_in,W -> bf16 row-major, U -> U^T bf16.
__global__ void node_prep_kernel(const float* __restrict__ Win,
                                 const float* __restrict__ W,
                                 const float* __restrict__ U,
                                 unsigned short* __restrict__ Winb,
                                 unsigned short* __restrict__ Wb,
                                 unsigned short* __restrict__ Utb) {
  int i = blockIdx.x * blockDim.x + threadIdx.x;
  if (i < MDIM * INDIM) Winb[i] = f32_bf16(Win[i]);
  if (i < MDIM * MDIM) {
    Wb[i] = f32_bf16(W[i]);
    int r = i >> 7, c = i & 127;
    Utb[c * MDIM + r] = f32_bf16(U[i]);
  }
}

__global__ __launch_bounds__(32 * NWAVES)
void node_main_kernel(const float* __restrict__ x, const float* __restrict__ b,
                      const float* __restrict__ Wcls, const float* __restrict__ bcls,
                      const unsigned short* __restrict__ Winb,
                      const unsigned short* __restrict__ Wb,
                      const unsigned short* __restrict__ Utb,
                      float* __restrict__ probs, float* __restrict__ xtr) {
  __shared__ unsigned short Abuf0[16 * ASTRIDE];  // layer input (bf16)
  __shared__ unsigned short Abuf1[16 * ASTRIDE];  // tanh hidden (bf16)
  __shared__ float clsbuf[NWAVES * 16];

  const int tid  = threadIdx.x;
  const int lane = tid & 31;
  const int w    = tid >> 5;            // wave id: owns N-tiles 2w, 2w+1
  const int n = lane & 15, g = lane >> 4;
  const int tb = w * TPW;               // first owned N-tile
  const int row0 = blockIdx.x * 16;

  // Per-lane bias / classifier weights for owned tiles (feature = 16*(tb+j)+n).
  float bv[TPW], wc[TPW];
#pragma unroll
  for (int j = 0; j < TPW; ++j) {
    bv[j] = b[(tb + j) * 16 + n];
    wc[j] = Wcls[(tb + j) * 16 + n];
  }

  // Per-lane output base pointer for x_transformed (B, M, T+1), time innermost.
  // Element (j, v) at time t lives at xp[v*VS + j*JS + t]; advance xp by 1/step.
  const int VS = MDIM * (TSTEPS + 1);   // batch-row stride  (6272 floats)
  const int JS = 16 * (TSTEPS + 1);     // N-tile stride     (784 floats)
  float* xp = xtr + ((size_t)(row0 + 8 * g) * MDIM + (tb * 16 + n)) * (TSTEPS + 1);

  // Stage this block's 16x64 slice of x into Abuf0 as bf16 (coalesced reads).
#pragma unroll
  for (int i = 0; i < 8; ++i) {
    int idx = i * 128 + tid;            // 0..1023
    int r = idx >> 6, c = idx & 63;
    Abuf0[r * ASTRIDE + c] = f32_bf16(x[(size_t)(row0 + r) * INDIM + c]);
  }
  lds_barrier();

  v8f acc[TPW];
  gemm_tiles<2, INDIM>(acc, Abuf0, Winb, tb, n, g);   // y0 tiles = x @ W_in^T

  v8f y[TPW];
#pragma unroll
  for (int j = 0; j < TPW; ++j) y[j] = acc[j];

  // x_transformed[:, :, 0] = y0
#pragma unroll
  for (int j = 0; j < TPW; ++j)
#pragma unroll
    for (int v = 0; v < 8; ++v) xp[v * VS + j * JS] = y[j][v];

  lds_barrier();     // all waves done reading x from Abuf0 -> safe to overwrite
#pragma unroll
  for (int j = 0; j < TPW; ++j)        // first RK input = y0 (own columns)
#pragma unroll
    for (int v = 0; v < 8; ++v)
      Abuf0[(v + 8 * g) * ASTRIDE + (tb + j) * 16 + n] = f32_bf16(y[j][v]);

  const float wadd[4] = {HSTEP / 6.f, HSTEP / 3.f, HSTEP / 3.f, HSTEP / 6.f};
  const float cnx[3]  = {HSTEP * 0.5f, HSTEP * 0.5f, HSTEP};

  for (int step = 0; step < TSTEPS; ++step) {
    v8f ynext[TPW];
#pragma unroll
    for (int j = 0; j < TPW; ++j) ynext[j] = y[j];

#pragma unroll
    for (int e = 0; e < 4; ++e) {
      lds_barrier();                                    // Abuf0 (input) ready
      gemm_tiles<4, MDIM>(acc, Abuf0, Wb, tb, n, g);    // in @ W^T (own tiles)

#pragma unroll
      for (int j = 0; j < TPW; ++j)                     // h = tanh(.+b) -> Abuf1
#pragma unroll
        for (int v = 0; v < 8; ++v)
          Abuf1[(v + 8 * g) * ASTRIDE + (tb + j) * 16 + n] =
              f32_bf16(fast_tanh(acc[j][v] + bv[j]));

      lds_barrier();         // Abuf1 ready; everyone done reading Abuf0
      gemm_tiles<4, MDIM>(acc, Abuf1, Utb, tb, n, g);   // k_e = h @ U

#pragma unroll
      for (int j = 0; j < TPW; ++j)                     // RK accumulate
#pragma unroll
        for (int v = 0; v < 8; ++v)
          ynext[j][v] += wadd[e] * acc[j][v];

      if (e < 3) {                                      // next eval input
#pragma unroll
        for (int j = 0; j < TPW; ++j)
#pragma unroll
          for (int v = 0; v < 8; ++v)
            Abuf0[(v + 8 * g) * ASTRIDE + (tb + j) * 16 + n] =
                f32_bf16(y[j][v] + cnx[e] * acc[j][v]);
      } else {
#pragma unroll
        for (int j = 0; j < TPW; ++j) y[j] = ynext[j];
#pragma unroll
        for (int j = 0; j < TPW; ++j)                   // next step input = y_{n+1}
#pragma unroll
          for (int v = 0; v < 8; ++v)
            Abuf0[(v + 8 * g) * ASTRIDE + (tb + j) * 16 + n] = f32_bf16(y[j][v]);
      }
    }

    // x_transformed[:, :, step+1] (own feature columns); background stores.
    xp += 1;
#pragma unroll
    for (int j = 0; j < TPW; ++j)
#pragma unroll
      for (int v = 0; v < 8; ++v) xp[v * VS + j * JS] = y[j][v];
  }

  // Classifier: per-wave partial dot over its 32 features, shfl-reduce over
  // the 16 lanes of each half, then cross-wave combine through LDS.
  float part[8];
#pragma unroll
  for (int v = 0; v < 8; ++v) {
    float s = 0.f;
#pragma unroll
    for (int j = 0; j < TPW; ++j) s += y[j][v] * wc[j];
    part[v] = s;
  }
#pragma unroll
  for (int m = 1; m < 16; m <<= 1)
#pragma unroll
    for (int v = 0; v < 8; ++v) part[v] += __shfl_xor(part[v], m, 32);

  if (n == 0) {
#pragma unroll
    for (int v = 0; v < 8; ++v) clsbuf[w * 16 + 8 * g + v] = part[v];
  }
  lds_barrier();
  if (tid < 16) {
    float s = clsbuf[tid] + clsbuf[16 + tid] + clsbuf[32 + tid] + clsbuf[48 + tid] +
              bcls[0];
    probs[row0 + tid] = 1.f / (1.f + expf(-s));
  }
}

extern "C" void kernel_launch(void* const* d_in, const int* in_sizes, int n_in,
                              void* d_out, int out_size, void* d_ws, size_t ws_size,
                              hipStream_t stream) {
  const float* x    = (const float*)d_in[0];
  const float* Win  = (const float*)d_in[1];
  const float* W    = (const float*)d_in[2];
  const float* b    = (const float*)d_in[3];
  const float* U    = (const float*)d_in[4];
  const float* Wcls = (const float*)d_in[5];
  const float* bcls = (const float*)d_in[6];

  unsigned short* Winb = (unsigned short*)d_ws;            // 128*64
  unsigned short* Wb   = Winb + MDIM * INDIM;              // 128*128
  unsigned short* Utb  = Wb + MDIM * MDIM;                 // 128*128 (U^T)

  float* probs = (float*)d_out;                            // (8192,)
  float* xtr   = probs + BATCH;                            // (8192,128,49)

  node_prep_kernel<<<64, 256, 0, stream>>>(Win, W, U, Winb, Wb, Utb);
  node_main_kernel<<<BATCH / 16, 32 * NWAVES, 0, stream>>>(x, b, Wcls, bcls, Winb,
                                                           Wb, Utb, probs, xtr);
}
